// PointNet_39479339385383
// MI455X (gfx1250) — compile-verified, matrix operand order checked by reading, not run on hardware
//
#include <hip/hip_runtime.h>

// ---------------------------------------------------------------------------
// PointNetConv x3 for MI455X (gfx1250): wave32 WMMA f16 (f32 accumulate).
// One wave owns 32 edges (or 32 nodes) = two 16-row WMMA tiles (MR=2).
// A fragments are hoisted into registers once per stage, so each stage runs
// IN PLACE in a single 16KB LDS buffer (DS ops are in-order within a wave).
// Each B fragment feeds two WMMAs -> ~1 memory op per WMMA.
// ---------------------------------------------------------------------------

typedef _Float16 v16h __attribute__((ext_vector_type(16)));
typedef _Float16 v8h  __attribute__((ext_vector_type(8)));
typedef _Float16 v4h  __attribute__((ext_vector_type(4)));
typedef float    v8f  __attribute__((ext_vector_type(8)));
typedef float    v4f  __attribute__((ext_vector_type(4)));
typedef unsigned v4u  __attribute__((ext_vector_type(4)));

#define NNODES 10000
#define NEDGES 320000
#define NEGINF_KEY 0x007FFFFFu   // monotone key of -inf

// Monotone order-preserving float->uint key (max over keys == max over floats)
__device__ inline unsigned fkey(float f) {
    unsigned u = __float_as_uint(f);
    return (u & 0x80000000u) ? ~u : (u | 0x80000000u);
}
__device__ inline float fdecode(unsigned k) {
    unsigned u = (k & 0x80000000u) ? (k ^ 0x80000000u) : ~k;
    return __uint_as_float(u);
}

// Assemble a v16h fragment from two 8-half contiguous loads.
__device__ inline v16h load_frag(const _Float16* p0, const _Float16* p1) {
    v8h lo = *(const v8h*)p0;
    v8h hh = *(const v8h*)p1;
    return __builtin_shufflevector(lo, hh, 0,1,2,3,4,5,6,7,8,9,10,11,12,13,14,15);
}

// One GEMM stage: [MR*16 x (NK*32)] (A, row-major f16 LDS, row stride 256)
//  x [(NK*32) x (NT*16)] (B, pre-packed lane layout in global) -> relu? -> LDS.
// A fragments are hoisted into registers up front, so outB may equal Ain.
template <int NK, int NT, int MR>
__device__ inline void gemm_stage_t(const _Float16* Ain, const _Float16* w,
                                    const float* bias, _Float16* outB,
                                    bool relu, int lane, int hi) {
    const int row = lane & 15;
    v16h a[MR][NK];
#pragma unroll
    for (int m = 0; m < MR; ++m)
#pragma unroll
        for (int kc = 0; kc < NK; ++kc) {
            const _Float16* rp = Ain + (m * 16 + row) * 256 + kc * 32 + hi * 8;
            a[m][kc] = load_frag(rp, rp + 16);
        }
#pragma unroll 2
    for (int nt = 0; nt < NT; ++nt) {
        v8f acc[MR];
#pragma unroll
        for (int m = 0; m < MR; ++m) acc[m] = (v8f){};
#pragma unroll
        for (int kc = 0; kc < NK; ++kc) {
            const _Float16* bp = w + (((size_t)(nt * NK + kc) * 32 + lane) << 4);
            v16h b = load_frag(bp, bp + 8);
#pragma unroll
            for (int m = 0; m < MR; ++m)
                acc[m] = __builtin_amdgcn_wmma_f32_16x16x32_f16(
                    false, a[m][kc], false, b, (short)0, acc[m], false, false);
        }
        int col = nt * 16 + row;
        float bv = bias[col];
#pragma unroll
        for (int m = 0; m < MR; ++m)
#pragma unroll
            for (int v = 0; v < 8; ++v) {
                float x = acc[m][v] + bv;
                if (relu) x = fmaxf(x, 0.f);
                outB[(m * 16 + v + hi * 8) * 256 + col] = (_Float16)x;
            }
    }
}

// --------------------------- weight packing --------------------------------
// dst[((nt*nk+kc)*32+lane)*16 + h] = W[k, n] (f16), k per ISA B-layout:
// lanes 0-15 hold K {0..7,16..23}, lanes 16-31 hold K {8..15,24..31}.
__global__ void pack_w_kernel(const float* __restrict__ W, _Float16* __restrict__ dst,
                              int K, int nk, int Ncols, int ntiles) {
    int idx = blockIdx.x * blockDim.x + threadIdx.x;
    int total = ntiles * nk * 32;
    if (idx >= total) return;
    int lane = idx & 31;
    int t = idx >> 5;
    int kc = t % nk;
    int nt = t / nk;
    int hi = lane >> 4;
    int n = nt * 16 + (lane & 15);
    _Float16* d = dst + ((size_t)idx << 4);
#pragma unroll
    for (int h = 0; h < 16; ++h) {
        int k = kc * 32 + ((h < 8) ? (hi * 8 + h) : (16 + hi * 8 + (h - 8)));
        float v = (k < K && n < Ncols) ? W[(size_t)k * Ncols + n] : 0.f;
        d[h] = (_Float16)v;
    }
}

__global__ void fill_u32_kernel(unsigned* __restrict__ p, unsigned v, int n) {
    int i = blockIdx.x * blockDim.x + threadIdx.x;
    if (i < n) p[i] = v;
}

// --------------------------- per-edge local MLP ----------------------------
template <int NK0>
__global__ __launch_bounds__(32)
void edge_conv_kernel(const float* __restrict__ feat,
                      const float* __restrict__ pos, const int* __restrict__ ei,
                      const _Float16* __restrict__ w0, const float* __restrict__ b0,
                      const _Float16* __restrict__ w1, const float* __restrict__ b1,
                      const _Float16* __restrict__ w2, const float* __restrict__ b2,
                      unsigned* __restrict__ aggKey) {
    __shared__ __align__(16) _Float16 sA[32 * 256];
    __shared__ int sSrc[32];
    __shared__ int sDst[32];

    const int lane = threadIdx.x & 31;
    const int hi = lane >> 4;
    const int EE = NEDGES + NNODES;
    const int base = blockIdx.x * 32;

    // One edge per lane: indices (self-loops synthesized for e >= NEDGES).
    {
        int e = base + lane;
        int s = 0, d = -1;
        if (e < EE) {
            if (e < NEDGES) { s = ei[e]; d = ei[NEDGES + e]; }
            else            { s = e - NEDGES; d = s; }
        }
        sSrc[lane] = s;
        sDst[lane] = d;
    }
    __syncthreads();

    // Gather [x_src, pos_src - pos_dst], f16, zero-padded to NK0*32 columns.
    if constexpr (NK0 == 1) {
        // Layer 0: featDim == 8, one row per lane, fully unrolled.
        const int r = lane;
        int s = sSrc[r];
        int d = sDst[r]; if (d < 0) d = s;
        v4f f0 = *(const v4f*)(feat + (size_t)s * 8);
        v4f f1 = *(const v4f*)(feat + (size_t)s * 8 + 4);
        *(v4h*)(sA + r * 256 + 0) = __builtin_convertvector(f0, v4h);
        *(v4h*)(sA + r * 256 + 4) = __builtin_convertvector(f1, v4h);
        sA[r * 256 + 8] = (_Float16)(pos[s * 2 + 0] - pos[d * 2 + 0]);
        sA[r * 256 + 9] = (_Float16)(pos[s * 2 + 1] - pos[d * 2 + 1]);
#pragma unroll
        for (int c = 10; c < 32; c += 2)
            *(unsigned*)(sA + r * 256 + c) = 0u;
    } else {
        // Layers 1/2: featDim == 128, vectorized float4 -> half4.
        for (int it = lane; it < 32 * 32; it += 32) {
            int r = it >> 5, q = it & 31;
            int s = sSrc[r];
            v4f f = *(const v4f*)(feat + (size_t)s * 128 + q * 4);
            *(v4h*)(sA + r * 256 + q * 4) = __builtin_convertvector(f, v4h);
        }
        const int r = lane;
        int s = sSrc[r];
        int d = sDst[r]; if (d < 0) d = s;
        sA[r * 256 + 128] = (_Float16)(pos[s * 2 + 0] - pos[d * 2 + 0]);
        sA[r * 256 + 129] = (_Float16)(pos[s * 2 + 1] - pos[d * 2 + 1]);
#pragma unroll
        for (int c = 130; c < 160; c += 2)
            *(unsigned*)(sA + r * 256 + c) = 0u;
    }

    gemm_stage_t<NK0, 16, 2>(sA, w0, b0, sA, true, lane, hi);  // K   -> 256
    gemm_stage_t<8,   16, 2>(sA, w1, b1, sA, true, lane, hi);  // 256 -> 256

    // 256 -> 128, then per-element segment max via monotone-key u32 atomics.
    const int row = lane & 15;
    v16h a[2][8];
#pragma unroll
    for (int m = 0; m < 2; ++m)
#pragma unroll
        for (int kc = 0; kc < 8; ++kc) {
            const _Float16* rp = sA + (m * 16 + row) * 256 + kc * 32 + hi * 8;
            a[m][kc] = load_frag(rp, rp + 16);
        }
#pragma unroll 2
    for (int nt = 0; nt < 8; ++nt) {
        v8f acc[2] = {(v8f){}, (v8f){}};
#pragma unroll
        for (int kc = 0; kc < 8; ++kc) {
            const _Float16* bp = w2 + (((size_t)(nt * 8 + kc) * 32 + lane) << 4);
            v16h b = load_frag(bp, bp + 8);
#pragma unroll
            for (int m = 0; m < 2; ++m)
                acc[m] = __builtin_amdgcn_wmma_f32_16x16x32_f16(
                    false, a[m][kc], false, b, (short)0, acc[m], false, false);
        }
        int col = nt * 16 + row;
        float bv = b2[col];
#pragma unroll
        for (int m = 0; m < 2; ++m)
#pragma unroll
            for (int v = 0; v < 8; ++v) {
                float x = acc[m][v] + bv;
                int d = sDst[m * 16 + v + hi * 8];
                if (d >= 0)
                    atomicMax(&aggKey[(size_t)d * 128 + col], fkey(x));
            }
    }
}

// --------------------------- per-node global MLP ---------------------------
template <int NTOUT>
__global__ __launch_bounds__(32)
void node_mlp_kernel(const unsigned* __restrict__ aggKey,
                     const _Float16* __restrict__ w0, const float* __restrict__ b0,
                     const _Float16* __restrict__ w1, const float* __restrict__ b1,
                     const _Float16* __restrict__ w2, const float* __restrict__ b2,
                     int nOut, int outerRelu, float* __restrict__ out) {
    __shared__ __align__(16) _Float16 sA[32 * 256];

    const int lane = threadIdx.x & 31;
    const int hi = lane >> 4;
    const int base = blockIdx.x * 32;

    // Gather + decode aggregated features [32 x 128], vectorized.
    for (int it = lane; it < 32 * 32; it += 32) {
        int r = it >> 5, q = it & 31;
        int node = base + r;
        v4f f = (v4f){};
        if (node < NNODES) {
            v4u k = *(const v4u*)(aggKey + (size_t)node * 128 + q * 4);
            f.x = fdecode(k.x); f.y = fdecode(k.y);
            f.z = fdecode(k.z); f.w = fdecode(k.w);
        }
        *(v4h*)(sA + r * 256 + q * 4) = __builtin_convertvector(f, v4h);
    }

    gemm_stage_t<4, 16, 2>(sA, w0, b0, sA, true, lane, hi);    // 128 -> 256
    gemm_stage_t<8, 16, 2>(sA, w1, b1, sA, true, lane, hi);    // 256 -> 256

    // 256 -> nOut (128 or 1), optional outer relu, fp32 store.
    const int row = lane & 15;
    v16h a[2][8];
#pragma unroll
    for (int m = 0; m < 2; ++m)
#pragma unroll
        for (int kc = 0; kc < 8; ++kc) {
            const _Float16* rp = sA + (m * 16 + row) * 256 + kc * 32 + hi * 8;
            a[m][kc] = load_frag(rp, rp + 16);
        }
#pragma unroll 1
    for (int nt = 0; nt < NTOUT; ++nt) {
        v8f acc[2] = {(v8f){}, (v8f){}};
#pragma unroll
        for (int kc = 0; kc < 8; ++kc) {
            const _Float16* bp = w2 + (((size_t)(nt * 8 + kc) * 32 + lane) << 4);
            v16h b = load_frag(bp, bp + 8);
#pragma unroll
            for (int m = 0; m < 2; ++m)
                acc[m] = __builtin_amdgcn_wmma_f32_16x16x32_f16(
                    false, a[m][kc], false, b, (short)0, acc[m], false, false);
        }
        int col = nt * 16 + row;
        float bv = (col < nOut) ? b2[col] : 0.f;
#pragma unroll
        for (int m = 0; m < 2; ++m)
#pragma unroll
            for (int v = 0; v < 8; ++v) {
                float x = acc[m][v] + bv;
                if (outerRelu) x = fmaxf(x, 0.f);
                int node = base + m * 16 + v + hi * 8;
                if (node < NNODES && col < nOut)
                    out[(size_t)node * nOut + col] = x;
            }
    }
}

// --------------------------------------------------------------------------
extern "C" void kernel_launch(void* const* d_in, const int* in_sizes, int n_in,
                              void* d_out, int out_size, void* d_ws, size_t ws_size,
                              hipStream_t stream) {
    (void)in_sizes; (void)n_in; (void)out_size; (void)ws_size;

    const float* x   = (const float*)d_in[0];
    const float* pos = (const float*)d_in[1];
    const int*   ei  = (const int*)d_in[2];
    const float* P[36];
    for (int i = 0; i < 36; ++i) P[i] = (const float*)d_in[3 + i];

    char* ws = (char*)d_ws;
    size_t off = 0;
    auto salloc = [&](size_t bytes) {
        size_t o = off;
        off = (off + bytes + 255) & ~(size_t)255;
        return o;
    };

    const size_t featBytes = (size_t)NNODES * 128 * sizeof(float);
    unsigned* aggKey = (unsigned*)(ws + salloc(featBytes));
    float*    hA     = (float*)(ws + salloc(featBytes));
    float*    hB     = (float*)(ws + salloc(featBytes));

    // 18 packed weight matrices: 6 per conv layer.
    //   local:  W0[K_in,256], W1[256,256], W2[256,128]
    //   global: W0[128,256],  W1[256,256], W2[256, l<2?128:1]
    int       pkW[18], pkK[18], pkNk[18], pkN[18], pkNt[18];
    _Float16* pkPtr[18];
    int t = 0;
    for (int l = 0; l < 3; ++l) {
        int bi  = l * 12;
        int kin = (l == 0) ? 10 : 130;
        int nk0 = (l == 0) ? 1 : 5;
        int descW[6]  = { bi + 0, bi + 2, bi + 4, bi + 6, bi + 8, bi + 10 };
        int descK[6]  = { kin, 256, 256, 128, 256, 256 };
        int descNk[6] = { nk0,   8,   8,   4,   8,   8 };
        int descN[6]  = { 256, 256, 128, 256, 256, (l < 2) ? 128 : 1 };
        int descNt[6] = { 16,   16,   8,  16,  16, (l < 2) ? 8 : 1 };
        for (int j = 0; j < 6; ++j, ++t) {
            pkW[t]  = descW[j];
            pkK[t]  = descK[j];
            pkNk[t] = descNk[j];
            pkN[t]  = descN[j];
            pkNt[t] = descNt[j];
            size_t bytes = (size_t)descNt[j] * descNk[j] * 32 * 16 * sizeof(_Float16);
            pkPtr[t] = (_Float16*)(ws + salloc(bytes));
        }
    }

    // Pack all weights (f32 -> f16 in WMMA B-fragment lane layout).
    for (int i = 0; i < 18; ++i) {
        int total  = pkNt[i] * pkNk[i] * 32;
        int blocks = (total + 255) / 256;
        pack_w_kernel<<<blocks, 256, 0, stream>>>(P[pkW[i]], pkPtr[i],
                                                  pkK[i], pkNk[i], pkN[i], pkNt[i]);
    }

    const int EE       = NEDGES + NNODES;
    const int eBlocks  = (EE + 31) / 32;
    const int nBlocks  = (NNODES + 31) / 32;
    const int aggElems = NNODES * 128;
    const int fBlocks  = (aggElems + 255) / 256;

    for (int l = 0; l < 3; ++l) {
        int base = l * 6;                  // packed-descriptor base for this conv
        int pbi  = l * 12;                 // param index base (rel. to P)
        const float* feat   = (l == 0) ? x : (l == 1 ? hA : hB);
        float*       outPtr = (l == 0) ? hA : (l == 1 ? hB : (float*)d_out);
        int          nOut   = (l < 2) ? 128 : 1;
        int          oRelu  = (l < 2) ? 1 : 0;

        fill_u32_kernel<<<fBlocks, 256, 0, stream>>>(aggKey, NEGINF_KEY, aggElems);

        if (l == 0) {
            edge_conv_kernel<1><<<eBlocks, 32, 0, stream>>>(
                feat, pos, ei,
                pkPtr[base + 0], P[pbi + 1],
                pkPtr[base + 1], P[pbi + 3],
                pkPtr[base + 2], P[pbi + 5],
                aggKey);
        } else {
            edge_conv_kernel<5><<<eBlocks, 32, 0, stream>>>(
                feat, pos, ei,
                pkPtr[base + 0], P[pbi + 1],
                pkPtr[base + 1], P[pbi + 3],
                pkPtr[base + 2], P[pbi + 5],
                aggKey);
        }

        if (l < 2) {
            node_mlp_kernel<8><<<nBlocks, 32, 0, stream>>>(
                aggKey,
                pkPtr[base + 3], P[pbi + 7],
                pkPtr[base + 4], P[pbi + 9],
                pkPtr[base + 5], P[pbi + 11],
                nOut, oRelu, outPtr);
        } else {
            node_mlp_kernel<1><<<nBlocks, 32, 0, stream>>>(
                aggKey,
                pkPtr[base + 3], P[pbi + 7],
                pkPtr[base + 4], P[pbi + 9],
                pkPtr[base + 5], P[pbi + 11],
                nOut, oRelu, outPtr);
        }
    }
}
